// CapsPrimary2d_74448963109576
// MI455X (gfx1250) — matrix operand order repaired and checked by reading, not compiled, not run on hardware
//
#include <hip/hip_runtime.h>

// ---------------------------------------------------------------------------
// CapsPrimary2d for MI455X (gfx1250):
//   conv -> bf16 WMMA implicit-GEMM (M=256, N=25088, K=20736)
//   A-tile (weights) loaded via Tensor Data Mover (tensor_load_to_lds, with
//   hardware LDS row padding), B-tile im2col gathered with a precomputed
//   offset table (L2-resident), routing done one wave32 per pixel.
// ---------------------------------------------------------------------------

typedef __attribute__((ext_vector_type(16))) __bf16       v16bf;
typedef __attribute__((ext_vector_type(8)))  float        v8f;
typedef __attribute__((ext_vector_type(4)))  unsigned int u32x4;
typedef __attribute__((ext_vector_type(8)))  unsigned int u32x8;

#define B_       32
#define CIN_     256
#define H_       64
#define W_       64
#define KSZ      9
#define STRIDE_  2
#define CT_      32               // capsule types
#define OS0_     8                // pose dim
#define OH_      28
#define OW_      28
#define COUT_    256              // CT_*OS0_
#define KTOT_    (CIN_*KSZ*KSZ)   // 20736 = 324*64
#define NTOT_    (B_*OH_*OW_)     // 25088 = 196*128
#define PIX_     (OH_*OW_)        // 784
#define EPS_     1e-8f

#define BM 128
#define BN 128
#define BK 64                      // two 16x16x32 WMMA k-substeps per stage
#define LDS_PAD 4                  // 4 bf16 = 8B = 2 DWORDs (TDM pad_amount=1)

// ---------------------------------------------------------------------------
// Kernel 1a: pack fp32 conv weights -> bf16 (row-major [cout][cin*9*9] is
// exactly the GEMM A layout: k = ci*81 + kh*9 + kw).
// ---------------------------------------------------------------------------
__global__ void pack_w_bf16(const float* __restrict__ w,
                            __bf16* __restrict__ wb, int n) {
    int i = blockIdx.x * blockDim.x + threadIdx.x;
    if (i < n) wb[i] = (__bf16)w[i];
}

// ---------------------------------------------------------------------------
// Kernel 1b: im2col offset table offs[k] = ci*H*W + kh*W + kw.
// 83 KB -> lives in L2 (192 MB) for the whole conv; removes all div/mod and
// carry logic from the GEMM hot loop.
// ---------------------------------------------------------------------------
__global__ void build_im2col_offs(int* __restrict__ offs) {
    int k = blockIdx.x * blockDim.x + threadIdx.x;
    if (k < KTOT_) {
        int ci = k / (KSZ * KSZ);
        int r  = k - ci * (KSZ * KSZ);
        int kh = r / KSZ;
        int kw = r - kh * KSZ;
        offs[k] = ci * (H_ * W_) + kh * W_ + kw;
    }
}

// ---------------------------------------------------------------------------
// TDM: async-load one BM x BK bf16 weight tile into LDS with hardware row
// padding. Descriptor per CDNA5 ISA (2D tensor -> 2-group
// form). Issued by one wave; completion via TENSORcnt.
// ---------------------------------------------------------------------------
__device__ __forceinline__ void tdm_load_a_tile(const __bf16* gsrc,
                                                unsigned lds_byte_off) {
    unsigned long long ga = (unsigned long long)(uintptr_t)gsrc;
    u32x4 g0;
    g0.x = 1u;                                   // count=1, user descriptor
    g0.y = lds_byte_off;                         // lds_addr (bytes)
    g0.z = (unsigned)ga;                         // global_addr[31:0]
    g0.w = (unsigned)(ga >> 32) | 0x80000000u;   // global_addr[56:32] | type=2

    u32x8 g1;
    // data_size=2B (code 1) | pad_enable | pad_interval=32dw | pad_amount=2dw
    g1[0] = (1u << 16) | (1u << 20) | (4u << 22) | (1u << 25);
    g1[1] = (KTOT_ & 0xffffu) << 16;             // tensor_dim0[15:0]
    g1[2] = ((KTOT_ >> 16) & 0xffffu)            // tensor_dim0[31:16]
          | ((COUT_ & 0xffffu) << 16);           // tensor_dim1[15:0]
    g1[3] = ((unsigned)BK) << 16;                // tile_dim0 (dim1 hi16 = 0)
    g1[4] = (unsigned)BM;                        // tile_dim1 (tile_dim2 = 0)
    g1[5] = (unsigned)KTOT_;                     // tensor_dim0_stride[31:0]
    g1[6] = 0u;                                  // stride hi | dim1_stride lo
    g1[7] = 0u;
    asm volatile("tensor_load_to_lds %0, %1" :: "s"(g0), "s"(g1) : "memory");
}

// ---------------------------------------------------------------------------
// Kernel 2: implicit-im2col GEMM conv with v_wmma_f32_16x16x32_bf16.
// Block = 256 threads = 8 waves (2 M x 4 N), block tile 128x128, K-step 64.
// Each wave: 4x2 grid of 16x16 tiles, 16 WMMAs per stage.
// Output caps[n*256 + m] so one capsule's 8 pose dims are contiguous.
// ---------------------------------------------------------------------------
__global__ __launch_bounds__(256) void caps_conv_wmma(
    const float*  __restrict__ x,
    const __bf16* __restrict__ w_bf,
    const int*    __restrict__ offs,
    const float*  __restrict__ conv_b,
    float*        __restrict__ caps)
{
    __shared__ __bf16 Als[BM][BK + LDS_PAD];   // [m_local][k_local] (TDM-fed)
    __shared__ __bf16 Bls[BN][BK + LDS_PAD];   // [n_local][k_local] (im2col^T)

    const int tid  = threadIdx.x;
    const int lane = tid & 31;
    const int wave = tid >> 5;
    const int wm0  = (wave >> 2) * 64;     // wave M offset in block tile
    const int wn0  = (wave & 3) * 32;      // wave N offset in block tile
    const int m0   = blockIdx.y * BM;      // 0 or 128
    const int n0   = blockIdx.x * BN;      // 0..25088-128

    // B-stage roles: 128 columns x 2 halves of 32 k's -> 256 threads
    const int bcol = tid & 127;
    const int bkh  = (tid >> 7) * 32;

    // im2col n-decomposition (fixed per thread; all dims exactly divisible)
    const int n_im = n0 + bcol;
    const int bb   = n_im / PIX_;
    const int p_im = n_im - bb * PIX_;
    const int oh   = p_im / OW_;
    const int ow   = p_im - oh * OW_;
    const float* xb = x + (size_t)bb * CIN_ * H_ * W_
                        + (size_t)(oh * STRIDE_) * W_ + (ow * STRIDE_);

    const unsigned alds = (unsigned)(uintptr_t)(&Als[0][0]); // LDS byte offset
    const __bf16*  wsrc = w_bf + (size_t)m0 * KTOT_;

    const v8f vzero = {0.f,0.f,0.f,0.f,0.f,0.f,0.f,0.f};
    v8f acc[4][2];
#pragma unroll
    for (int mt = 0; mt < 4; ++mt)
#pragma unroll
        for (int nt = 0; nt < 2; ++nt) acc[mt][nt] = vzero;

    const int lrow  = lane & 15;          // fragment row/col selector
    const int khalf = lane >> 4;          // 0: low K half, 1: high K half

    for (int k0 = 0; k0 < KTOT_; k0 += BK) {
        // ---- A tile: one async TDM DMA per stage (wave 0 issues) ----------
        if (wave == 0)
            tdm_load_a_tile(wsrc + k0, alds);

        // ---- B tile: im2col gather via offset table, f32 -> bf16 ----------
        {
            const int* op = offs + k0 + bkh;       // 32 contiguous offsets
            __builtin_prefetch(op + BK, 0, 1);     // next stage's offsets
#pragma unroll
            for (int j = 0; j < 32; ++j)
                Bls[bcol][bkh + j] = (__bf16)xb[op[j]];
        }

        if (wave == 0)
            __builtin_amdgcn_s_wait_tensorcnt(0);  // A tile landed in LDS
        __syncthreads();

        // ---- two 16x16x32 k-substeps, frag regs reused --------------------
#pragma unroll
        for (int ks = 0; ks < BK; ks += 32) {
            // A 16x32 frag: lane<16 -> M=lane, K {0..7,16..23};
            //               lane>=16 -> same M, K {8..15,24..31}
            v16bf afr[4];
#pragma unroll
            for (int mt = 0; mt < 4; ++mt) {
                const __bf16* ap = &Als[wm0 + mt * 16 + lrow][ks];
#pragma unroll
                for (int e = 0; e < 8; ++e) afr[mt][e]     = ap[khalf * 8 + e];
#pragma unroll
                for (int e = 0; e < 8; ++e) afr[mt][8 + e] = ap[16 + khalf * 8 + e];
            }
            // B 32x16 frag: column N = lane&15; lane<16: K {0..7,16..23}, else {8..15,24..31}
            v16bf bfr[2];
#pragma unroll
            for (int nt = 0; nt < 2; ++nt) {
                const __bf16* bp = &Bls[wn0 + nt * 16 + lrow][ks];
#pragma unroll
                for (int e = 0; e < 8; ++e) bfr[nt][e]     = bp[khalf * 8 + e];
#pragma unroll
                for (int e = 0; e < 8; ++e) bfr[nt][8 + e] = bp[16 + khalf * 8 + e];
            }
#pragma unroll
            for (int mt = 0; mt < 4; ++mt)
#pragma unroll
                for (int nt = 0; nt < 2; ++nt)
                    acc[mt][nt] = __builtin_amdgcn_wmma_f32_16x16x32_bf16(
                        false, afr[mt], false, bfr[nt],
                        (short)0, acc[mt][nt], false, false);
        }
        __syncthreads();
    }

    // ---- epilogue: C/D layout -> caps[n*256 + m] (+bias) ------------------
    const int mhalf = khalf * 8;   // VGPR i: M = i (lanes 0-15) / i+8 (16-31)
#pragma unroll
    for (int mt = 0; mt < 4; ++mt)
#pragma unroll
        for (int nt = 0; nt < 2; ++nt) {
            const int nn = n0 + wn0 + nt * 16 + lrow;
#pragma unroll
            for (int i = 0; i < 8; ++i) {
                const int mm = m0 + wm0 + mt * 16 + mhalf + i;
                caps[(size_t)nn * COUT_ + mm] = acc[mt][nt][i] + conv_b[mm];
            }
        }
}

// ---------------------------------------------------------------------------
// Kernel 3: squash + 3 routing iterations + activations.
// One wave32 per output pixel; lane = capsule type (exactly 32).
// Softmax over capsule types == cross-lane reduction (wave32 sweet spot).
// ---------------------------------------------------------------------------
__global__ __launch_bounds__(256) void caps_route(
    const float* __restrict__ caps,
    const float* __restrict__ rbias,
    float*       __restrict__ out)
{
    const int lane = threadIdx.x & 31;                 // capsule type c
    const int wave = threadIdx.x >> 5;
    const int pix  = blockIdx.x * 8 + wave;            // [0, 25088)
    const int bb   = pix / PIX_;
    const int p    = pix - bb * PIX_;

    float v[OS0_], pose[OS0_], rb[OS0_];
    const float* cp = caps  + (size_t)pix * COUT_ + lane * OS0_;
    const float* rp = rbias + ((size_t)lane * PIX_ + p) * OS0_;
#pragma unroll
    for (int i = 0; i < OS0_; ++i) { v[i] = cp[i]; rb[i] = rp[i]; }

    // votes = squash(caps)
    float n2 = 0.f;
#pragma unroll
    for (int i = 0; i < OS0_; ++i) n2 += v[i] * v[i];
    float sc = (n2 / (1.f + n2)) * rsqrtf(n2 + EPS_);
#pragma unroll
    for (int i = 0; i < OS0_; ++i) v[i] *= sc;

    float logit = 0.f;
#pragma unroll
    for (int it = 0; it < 3; ++it) {
        // softmax over the 32 capsule types (cross-lane)
        float mx = logit;
#pragma unroll
        for (int off = 16; off; off >>= 1)
            mx = fmaxf(mx, __shfl_xor(mx, off, 32));
        float e = __expf(logit - mx);
        float s = e;
#pragma unroll
        for (int off = 16; off; off >>= 1)
            s += __shfl_xor(s, off, 32);
        const float coup = e / s;

        float n2s = 0.f;
#pragma unroll
        for (int i = 0; i < OS0_; ++i) {
            pose[i] = coup * v[i] + rb[i];
            n2s += pose[i] * pose[i];
        }
        const float sc2 = (n2s / (1.f + n2s)) * rsqrtf(n2s + EPS_);
        float dot = 0.f;
#pragma unroll
        for (int i = 0; i < OS0_; ++i) {
            pose[i] *= sc2;
            dot += v[i] * pose[i];
        }
        logit += dot;
    }

    float a2 = 0.f;
#pragma unroll
    for (int i = 0; i < OS0_; ++i) a2 += pose[i] * pose[i];
    const float act = sqrtf(a2);

    // poses: [b, C, oh, ow, 8, 1] then acts: [b, C, oh, ow]
    float* po = out + (((size_t)bb * CT_ + lane) * PIX_ + p) * OS0_;
#pragma unroll
    for (int i = 0; i < OS0_; ++i) po[i] = pose[i];
    out[(size_t)B_ * CT_ * PIX_ * OS0_ + ((size_t)bb * CT_ + lane) * PIX_ + p] = act;
}

// ---------------------------------------------------------------------------
extern "C" void kernel_launch(void* const* d_in, const int* in_sizes, int n_in,
                              void* d_out, int out_size, void* d_ws, size_t ws_size,
                              hipStream_t stream) {
    const float* x      = (const float*)d_in[0];   // [32,256,64,64]
    const float* conv_w = (const float*)d_in[1];   // [256,256,9,9]
    const float* conv_b = (const float*)d_in[2];   // [256]
    const float* rbias  = (const float*)d_in[3];   // [32,28,28,8,1]
    float* out = (float*)d_out;

    // workspace: bf16 weights (10.6 MB) | offs table (83 KB) | caps (25.7 MB)
    __bf16* w_bf = (__bf16*)d_ws;
    int*    offs = (int*)((char*)d_ws + (size_t)COUT_ * KTOT_ * sizeof(__bf16));
    float*  caps = (float*)((char*)offs + (size_t)KTOT_ * sizeof(int));

    const int nw = COUT_ * KTOT_;                       // 5,308,416
    pack_w_bf16<<<(nw + 255) / 256, 256, 0, stream>>>(conv_w, w_bf, nw);
    build_im2col_offs<<<(KTOT_ + 255) / 256, 256, 0, stream>>>(offs);

    dim3 grid(NTOT_ / BN, COUT_ / BM);                  // (196, 2)
    caps_conv_wmma<<<grid, 256, 0, stream>>>(x, w_bf, offs, conv_b, caps);

    caps_route<<<NTOT_ / 8, 256, 0, stream>>>(caps, rbias, out);
}
